// GRU_90340342104500
// MI455X (gfx1250) — compile-verified, matrix operand order checked by reading, not run on hardware
//
#include <hip/hip_runtime.h>
#include <hip/hip_bf16.h>

// Problem constants (match reference)
#define B_   64
#define T_   2048
#define H_   128
#define G3H  384     // 3*H

typedef __bf16 bf16;
typedef __attribute__((ext_vector_type(16))) __bf16 bf16x16;
typedef __attribute__((ext_vector_type(8)))  __bf16 bf16x8;
typedef __attribute__((ext_vector_type(8)))  float  f32x8;

// Load a 16x32 bf16 WMMA A/B tile from a row-major [rows x ldk] bf16 array in LDS.
// CDNA5 wave32 layout (ISA 7.12.2): lanes 0-15 hold row (lane), K = {k0..k0+7, k0+16..k0+23};
// lanes 16-31 hold row (lane-16), K = {k0+8..k0+15, k0+24..k0+31}.
// Both halves are two contiguous 16B chunks -> two ds_load_b128 per lane.
__device__ __forceinline__ bf16x16 ld_tile(const bf16* base, int row0, int ldk, int k0, int lane) {
    int r  = row0 + (lane & 15);
    int kq = k0 + ((lane >> 4) << 3);     // +0 for lanes 0-15, +8 for lanes 16-31
    const bf16* p = base + r * ldk + kq;
    bf16x8 lo = *(const bf16x8*)(p);
    bf16x8 hi = *(const bf16x8*)(p + 16);
    return __builtin_shufflevector(lo, hi, 0,1,2,3,4,5,6,7,8,9,10,11,12,13,14,15);
}

__device__ __forceinline__ f32x8 wmma_bf16(bf16x16 a, bf16x16 b, f32x8 c) {
    // (neg_a, A, neg_b, B, c_mod, C, reuse_a, reuse_b)
    return __builtin_amdgcn_wmma_f32_16x16x32_bf16(false, a, false, b, (short)0, c, false, false);
}

__device__ __forceinline__ float sigmoidf_(float x) { return 1.0f / (1.0f + __expf(-x)); }

// Persistent single-WGP GRU layer scan.
// 1024 threads = 32 waves. Wave w: mi = w>>3 (batch tile 0..3), nj = w&7 (hidden tile 0..7).
// Each wave owns the (16 batch x 16 hidden) patch for ALL THREE gates -> gate math in registers.
// All weights live in LDS as bf16; h kept in LDS as fp32 (state) + bf16 (WMMA operand).
template<int CIN, bool IN_BF16, bool WRITE_Y>
__global__ __launch_bounds__(1024)
void gru_scan_kernel(const void* __restrict__ xin,
                     const float* __restrict__ W_ih, const float* __restrict__ W_hh,
                     const float* __restrict__ b_ih, const float* __restrict__ b_hh,
                     bf16* __restrict__ y_out,          // [T][B][H] bf16 (layer-0 only)
                     float* __restrict__ hT_out)        // [B][H] fp32
{
    extern __shared__ char smem_raw[];
    bf16*  Whh  = (bf16*)smem_raw;            // [384][128]
    bf16*  Wih  = Whh  + G3H * H_;            // [384][CIN]
    bf16*  hbf  = Wih  + G3H * CIN;           // [64][128]  h as bf16 (WMMA A)
    bf16*  xbf  = hbf  + B_  * H_;            // [64][CIN]  x_t as bf16 (WMMA A)
    float* hf   = (float*)(xbf + B_ * CIN);   // [64][128]  h as fp32 (state)
    float* bi_s = hf   + B_  * H_;            // [384]
    float* bh_s = bi_s + G3H;                 // [384]

    const int tid  = threadIdx.x;
    const int lane = tid & 31;
    const int wave = tid >> 5;
    const int bi   = (wave >> 3) * 16;        // batch tile row base
    const int j0   = (wave & 7) * 16;         // hidden tile col base

    // ---- init: weights -> LDS (fp32 -> bf16), biases, h = 0
    for (int i = tid; i < G3H * H_;  i += 1024) Whh[i] = (bf16)W_hh[i];
    for (int i = tid; i < G3H * CIN; i += 1024) Wih[i] = (bf16)W_ih[i];
    for (int i = tid; i < G3H;       i += 1024) { bi_s[i] = b_ih[i]; bh_s[i] = b_hh[i]; }
    for (int i = tid; i < B_ * H_;   i += 1024) { hf[i] = 0.0f; hbf[i] = (bf16)0.0f; }
    __syncthreads();

    const int col  = j0 + (lane & 15);           // hidden index j for this lane
    const int row8 = bi + ((lane >> 4) << 3);    // batch row base for acc element v
    const float brz_r = bi_s[col]        + bh_s[col];
    const float brz_z = bi_s[H_ + col]   + bh_s[H_ + col];
    const float b_in  = bi_s[2*H_ + col];
    const float b_hn  = bh_s[2*H_ + col];

    for (int t = 0; t < T_; ++t) {
        // ---- stage x_t into LDS (bf16)
        if (IN_BF16) {
            const bf16* xs = ((const bf16*)xin) + (size_t)t * (B_ * CIN);
            for (int i = tid; i < B_ * CIN; i += 1024) xbf[i] = xs[i];
        } else {
            const float* xs = (const float*)xin;   // [B][T][CIN]
            for (int i = tid; i < B_ * CIN; i += 1024) {
                int b = i / CIN, c = i % CIN;
                xbf[i] = (bf16)xs[((size_t)b * T_ + t) * CIN + c];
            }
            // prefetch next timestep's rows (global_prefetch_b8)
            if (tid < B_ && t + 1 < T_)
                __builtin_prefetch((const float*)xin + ((size_t)tid * T_ + (t + 1)) * CIN, 0, 1);
        }
        __syncthreads();

        // ---- gate pre-activations via WMMA (fp32 accumulate)
        f32x8 acc_r = {}; f32x8 acc_z = {}; f32x8 a_gh = {}; f32x8 a_gx = {};
        #pragma unroll
        for (int k0 = 0; k0 < H_; k0 += 32) {            // h @ W_hh^T  (K = 128)
            bf16x16 a = ld_tile(hbf, bi, H_, k0, lane);
            acc_r = wmma_bf16(a, ld_tile(Whh,          j0, H_, k0, lane), acc_r);
            acc_z = wmma_bf16(a, ld_tile(Whh, H_   +   j0, H_, k0, lane), acc_z);
            a_gh  = wmma_bf16(a, ld_tile(Whh, 2*H_ +   j0, H_, k0, lane), a_gh);
        }
        #pragma unroll
        for (int k0 = 0; k0 < CIN; k0 += 32) {           // x_t @ W_ih^T (K = CIN)
            bf16x16 a = ld_tile(xbf, bi, CIN, k0, lane);
            acc_r = wmma_bf16(a, ld_tile(Wih,          j0, CIN, k0, lane), acc_r);
            acc_z = wmma_bf16(a, ld_tile(Wih, H_   +   j0, CIN, k0, lane), acc_z);
            a_gx  = wmma_bf16(a, ld_tile(Wih, 2*H_ +   j0, CIN, k0, lane), a_gx);
        }

        // ---- gates + state update, all in registers (r,z,n share this lane's (b, j))
        float hnew[8];
        #pragma unroll
        for (int v = 0; v < 8; ++v) {
            int b = row8 + v;
            float r = sigmoidf_(acc_r[v] + brz_r);
            float z = sigmoidf_(acc_z[v] + brz_z);
            float n = tanhf(a_gx[v] + b_in + r * (a_gh[v] + b_hn));
            hnew[v] = (1.0f - z) * n + z * hf[b * H_ + col];
        }
        __syncthreads();   // all waves done reading hbf/hf/xbf before overwrite
        #pragma unroll
        for (int v = 0; v < 8; ++v) {
            int b = row8 + v;
            hf[b * H_ + col] = hnew[v];
            bf16 hb = (bf16)hnew[v];
            hbf[b * H_ + col] = hb;
            if (WRITE_Y) y_out[(size_t)t * (B_ * H_) + b * H_ + col] = hb;
        }
        __syncthreads();   // h_new visible before next step's reads
    }

    for (int i = tid; i < B_ * H_; i += 1024) hT_out[i] = hf[i];
}

// out[b][o] = concat(h0T, h1T)[b] . W_out[o] + b_out[o]   (64x10, trivially small)
__global__ void gru_head_kernel(const float* __restrict__ h0T, const float* __restrict__ h1T,
                                const float* __restrict__ W_out, const float* __restrict__ b_out,
                                float* __restrict__ out)
{
    int tid = blockIdx.x * blockDim.x + threadIdx.x;
    if (tid >= B_ * 10) return;
    int b = tid / 10, o = tid % 10;
    const float* w = W_out + o * (2 * H_);
    float acc = b_out[o];
    #pragma unroll 4
    for (int j = 0; j < H_; ++j) acc += h0T[b * H_ + j] * w[j];
    #pragma unroll 4
    for (int j = 0; j < H_; ++j) acc += h1T[b * H_ + j] * w[H_ + j];
    out[tid] = acc;
}

static inline size_t scan_smem_bytes(int cin) {
    size_t nbf = (size_t)G3H * H_ + (size_t)G3H * cin + (size_t)B_ * H_ + (size_t)B_ * cin;
    size_t nf  = (size_t)B_ * H_ + 2 * (size_t)G3H;
    return nbf * sizeof(bf16) + nf * sizeof(float);
}

extern "C" void kernel_launch(void* const* d_in, const int* in_sizes, int n_in,
                              void* d_out, int out_size, void* d_ws, size_t ws_size,
                              hipStream_t stream) {
    const float* x    = (const float*)d_in[0];
    const float* Wih0 = (const float*)d_in[1];
    const float* Whh0 = (const float*)d_in[2];
    const float* bih0 = (const float*)d_in[3];
    const float* bhh0 = (const float*)d_in[4];
    const float* Wih1 = (const float*)d_in[5];
    const float* Whh1 = (const float*)d_in[6];
    const float* bih1 = (const float*)d_in[7];
    const float* bhh1 = (const float*)d_in[8];
    const float* Wout = (const float*)d_in[9];
    const float* bout = (const float*)d_in[10];
    float* out = (float*)d_out;

    // workspace: y0 (bf16 [T][B][H], 32 MB) | h0T (fp32 8K) | h1T (fp32 8K)
    char*  ws  = (char*)d_ws;
    bf16*  y0  = (bf16*)ws;
    float* h0T = (float*)(ws + (size_t)T_ * B_ * H_ * sizeof(bf16));
    float* h1T = h0T + B_ * H_;

    const size_t smem0 = scan_smem_bytes(64);    // ~203 KB
    const size_t smem1 = scan_smem_bytes(128);   // ~259 KB (fits 320 KB WGP LDS)

    (void)hipFuncSetAttribute(reinterpret_cast<const void*>(&gru_scan_kernel<64, false, true>),
                              hipFuncAttributeMaxDynamicSharedMemorySize, (int)smem0);
    (void)hipFuncSetAttribute(reinterpret_cast<const void*>(&gru_scan_kernel<128, true, false>),
                              hipFuncAttributeMaxDynamicSharedMemorySize, (int)smem1);

    // Layer 0: x fp32 -> y0 bf16 + h0T
    hipLaunchKernelGGL((gru_scan_kernel<64, false, true>), dim3(1), dim3(1024), smem0, stream,
                       (const void*)x, Wih0, Whh0, bih0, bhh0, y0, h0T);
    // Layer 1: y0 bf16 -> h1T (y not needed)
    hipLaunchKernelGGL((gru_scan_kernel<128, true, false>), dim3(1), dim3(1024), smem1, stream,
                       (const void*)y0, Wih1, Whh1, bih1, bhh1, (bf16*)nullptr, h1T);
    // Head
    hipLaunchKernelGGL(gru_head_kernel, dim3(1), dim3(640), 0, stream, h0T, h1T, Wout, bout, out);
}
